// NBodyGNN_6914897347306
// MI455X (gfx1250) — matrix operand order repaired and checked by reading, not compiled, not run on hardware
//
#include <hip/hip_runtime.h>
#include <hip/hip_bf16.h>

#define N_BODIES 512
#define HID      128
#define IN_DIM   7
#define HEADS    2
#define OUT_DIM  (N_BODIES * 6)   // 3072
#define NEG_SLOPE 0.2f

typedef __attribute__((ext_vector_type(16))) __bf16 v16bf;
typedef __attribute__((ext_vector_type(8)))  float  v8f;
typedef __attribute__((ext_vector_type(4)))  int    v4i;

#if defined(__HIP_DEVICE_COMPILE__) && __has_builtin(__builtin_amdgcn_global_load_async_to_lds_b128) && __has_builtin(__builtin_amdgcn_s_wait_asynccnt)
#define HAVE_ASYNC_LDS 1
typedef __attribute__((address_space(1))) v4i* g4ptr_t;   // global v4i*
typedef __attribute__((address_space(3))) v4i* l4ptr_t;   // LDS v4i*
#else
#define HAVE_ASYNC_LDS 0
#endif

__device__ __forceinline__ __bf16 f2bf(float f) { return (__bf16)f; }
__device__ __forceinline__ float  lrelu(float v) { return v > 0.f ? v : NEG_SLOPE * v; }

// ---------------------------------------------------------------------------
// WMMA 16x16x32 bf16 fragment loaders (layouts per CDNA5 ISA §7.12.2)
// A (M=16 x K=32): lane L: m=L&15, g=L>>4; VGPR v holds K=(v&4?16:0)+g*8+(v&3)*2, +1
// B (K=32 x N=16): lane L: n=L&15, g=L>>4; VGPR v holds K=(v&4?16:0)+g*8+(v&3)*2, +1
// C/D (16x16 f32): elem r: row = r + 8*(L>>4), col = L&15
// ---------------------------------------------------------------------------
__device__ __forceinline__ v16bf frag_A_f32(const float* A, int lda, int k0, int lane) {
  const int m = lane & 15;
  const int g = lane >> 4;
  v16bf a;
#pragma unroll
  for (int v = 0; v < 8; ++v) {
    const int kb = k0 + ((v & 4) << 2) + g * 8 + ((v & 3) << 1);
    a[2 * v]     = f2bf(A[m * lda + kb]);
    a[2 * v + 1] = f2bf(A[m * lda + kb + 1]);
  }
  return a;
}

__device__ __forceinline__ v16bf frag_A_bf(const __bf16* A, int lda, int k0, int lane) {
  const int m = lane & 15;
  const int g = lane >> 4;
  v16bf a;
#pragma unroll
  for (int v = 0; v < 8; ++v) {
    const int kb = k0 + ((v & 4) << 2) + g * 8 + ((v & 3) << 1);
    a[2 * v]     = A[m * lda + kb];
    a[2 * v + 1] = A[m * lda + kb + 1];
  }
  return a;
}

__device__ __forceinline__ v16bf frag_B_f32(const float* B, int ldb, int k0, int n0, int lane) {
  const int n = n0 + (lane & 15);
  const int g = lane >> 4;
  v16bf b;
#pragma unroll
  for (int v = 0; v < 8; ++v) {
    const int k = k0 + ((v & 4) << 2) + g * 8 + ((v & 3) << 1);
    b[2 * v]     = f2bf(B[k * ldb + n]);
    b[2 * v + 1] = f2bf(B[(k + 1) * ldb + n]);
  }
  return b;
}

// Async (or fallback) copy of n16 16-byte chunks global -> LDS by 256 threads
__device__ __forceinline__ void stage_to_lds_16B(const void* gsrc, void* lds, int n16, int tid) {
#if HAVE_ASYNC_LDS
  char* gbase = const_cast<char*>((const char*)gsrc);
  for (int it = tid; it < n16; it += 256) {
    __builtin_amdgcn_global_load_async_to_lds_b128(
        (g4ptr_t)(gbase + 16 * it), (l4ptr_t)((char*)lds + 16 * it), 0, 0);
  }
  __builtin_amdgcn_s_wait_asynccnt(0);
#else
  const uint4* s = (const uint4*)gsrc;
  uint4* d = (uint4*)lds;
  for (int it = tid; it < n16; it += 256) d[it] = s[it];
#endif
  __syncthreads();
}

// ---------------------------------------------------------------------------
// Stage 0: Sx[c] = sum_i x[i][c]   (complete graph => segment_sum = colsum - self)
// ---------------------------------------------------------------------------
__global__ void k_colsum_x(const float* __restrict__ x, float* __restrict__ Sx) {
  __shared__ float red[256];
  const int c = blockIdx.x, t = threadIdx.x;
  red[t] = x[t * IN_DIM + c] + x[(t + 256) * IN_DIM + c];
  __syncthreads();
  for (int off = 128; off > 0; off >>= 1) {
    if (t < off) red[t] += red[t + off];
    __syncthreads();
  }
  if (t == 0) Sx[c] = red[0];
}

// Stage 1: h1 = relu( x @ (W_root1 + W_res - W_rel1) + Sx @ W_rel1 + b_rel1 + b_res )
__global__ void k_stage1(const float* __restrict__ x, const float* __restrict__ Sx,
                         const float* __restrict__ W_res, const float* __restrict__ b_res,
                         const float* __restrict__ W_rel1, const float* __restrict__ b_rel1,
                         const float* __restrict__ W_root1, float* __restrict__ h1) {
  const int i = blockIdx.x, j = threadIdx.x;
  __shared__ float xs[IN_DIM];
  if (j < IN_DIM) xs[j] = x[i * IN_DIM + j];
  __syncthreads();
  float acc = b_rel1[j] + b_res[j];
#pragma unroll
  for (int c = 0; c < IN_DIM; ++c) {
    const float wr = W_rel1[c * HID + j];
    acc += xs[c] * (W_root1[c * HID + j] + W_res[c * HID + j] - wr) + Sx[c] * wr;
  }
  h1[i * HID + j] = acc > 0.f ? acc : 0.f;
}

// Stage 2 prep: Sh = colsum(h1); base2[j] = Sh @ W_rel2 + b_rel2
__global__ void k_sh_base2(const float* __restrict__ h1, const float* __restrict__ W_rel2,
                           const float* __restrict__ b_rel2, float* __restrict__ base2) {
  __shared__ float Sh[HID];
  const int j = threadIdx.x;
  float s = 0.f;
  for (int i = 0; i < N_BODIES; ++i) s += h1[i * HID + j];
  Sh[j] = s;
  __syncthreads();
  float acc = b_rel2[j];
  for (int k = 0; k < HID; ++k) acc += Sh[k] * W_rel2[k * HID + j];
  base2[j] = acc;
}

// Stage 2: h2 = relu( h1 @ (W_root2 - W_rel2) + base2 )  [WMMA bf16, 512x128x128]
__global__ void k_gconv2(const float* __restrict__ h1, const float* __restrict__ Wrel2,
                         const float* __restrict__ Wroot2, const float* __restrict__ base2,
                         float* __restrict__ h2) {
  __shared__ float As[16 * HID];
  const int tid = threadIdx.x;
  const int m0 = blockIdx.x * 16;
  stage_to_lds_16B(h1 + (size_t)m0 * HID, As, 16 * HID * 4 / 16, tid);
  const int wave = tid >> 5, lane = tid & 31;
  const int n0 = wave * 16;
  v8f acc = {};
#pragma unroll
  for (int k0 = 0; k0 < HID; k0 += 32) {
    v16bf a = frag_A_f32(As, HID, k0, lane);
    v16bf b;
    {
      const int n = n0 + (lane & 15);
      const int g = lane >> 4;
#pragma unroll
      for (int v = 0; v < 8; ++v) {
        const int k = k0 + ((v & 4) << 2) + g * 8 + ((v & 3) << 1);
        b[2 * v]     = f2bf(Wroot2[k * HID + n] - Wrel2[k * HID + n]);
        b[2 * v + 1] = f2bf(Wroot2[(k + 1) * HID + n] - Wrel2[(k + 1) * HID + n]);
      }
    }
    acc = __builtin_amdgcn_wmma_f32_16x16x32_bf16(false, a, false, b, (short)0, acc, false, false);
  }
  const int g = lane >> 4;
  const int n = n0 + (lane & 15);
  const float bb = base2[n];
#pragma unroll
  for (int r = 0; r < 8; ++r) {
    const float v = acc[r] + bb;
    h2[(m0 + r + 8 * g) * HID + n] = v > 0.f ? v : 0.f;
  }
}

// GAT linear: xh = h2 @ W_gat  (512x128 @ 128x256)  [WMMA bf16]
__global__ void k_gat_lin(const float* __restrict__ h2, const float* __restrict__ Wgat,
                          float* __restrict__ xh) {
  __shared__ float As[16 * HID];
  const int tid = threadIdx.x;
  const int m0 = blockIdx.x * 16;
  stage_to_lds_16B(h2 + (size_t)m0 * HID, As, 16 * HID * 4 / 16, tid);
  const int wave = tid >> 5, lane = tid & 31;
  const int n0 = blockIdx.y * 128 + wave * 16;
  v8f acc = {};
#pragma unroll
  for (int k0 = 0; k0 < HID; k0 += 32) {
    v16bf a = frag_A_f32(As, HID, k0, lane);
    v16bf b = frag_B_f32(Wgat, 2 * HID, k0, n0, lane);
    acc = __builtin_amdgcn_wmma_f32_16x16x32_bf16(false, a, false, b, (short)0, acc, false, false);
  }
  const int g = lane >> 4, n = n0 + (lane & 15);
#pragma unroll
  for (int r = 0; r < 8; ++r)
    xh[(m0 + r + 8 * g) * (2 * HID) + n] = acc[r];
}

// Attention scores: a_s[i,hd] = xh[i,hd,:]·att_src[hd], a_d likewise
__global__ void k_att_scores(const float* __restrict__ xh, const float* __restrict__ att_src,
                             const float* __restrict__ att_dst,
                             float* __restrict__ a_s, float* __restrict__ a_d) {
  __shared__ float rs[HID], rd[HID];
  const int i = blockIdx.x, hd = blockIdx.y, c = threadIdx.x;
  const float xv = xh[i * (2 * HID) + hd * HID + c];
  rs[c] = xv * att_src[hd * HID + c];
  rd[c] = xv * att_dst[hd * HID + c];
  __syncthreads();
  for (int off = 64; off > 0; off >>= 1) {
    if (c < off) { rs[c] += rs[c + off]; rd[c] += rd[c + off]; }
    __syncthreads();
  }
  if (c == 0) { a_s[i * HEADS + hd] = rs[0]; a_d[i * HEADS + hd] = rd[0]; }
}

// amax[hd] = max_j a_s[j,hd]  (lrelu monotonic => segment-max collapses globally)
__global__ void k_asmax(const float* __restrict__ a_s, float* __restrict__ amax) {
  __shared__ float red[256];
  const int hd = blockIdx.x, t = threadIdx.x;
  red[t] = fmaxf(a_s[t * HEADS + hd], a_s[(t + 256) * HEADS + hd]);
  __syncthreads();
  for (int off = 128; off > 0; off >>= 1) {
    if (t < off) red[t] = fmaxf(red[t], red[t + off]);
    __syncthreads();
  }
  if (t == 0) amax[hd] = red[0];
}

// P[hd][i][j] = exp(lrelu(a_s[j]+a_d[i]) - lrelu(amax+a_d[i]))  stored bf16
__global__ void k_build_P(const float* __restrict__ a_s, const float* __restrict__ a_d,
                          const float* __restrict__ amax, __bf16* __restrict__ P) {
  const int gid = blockIdx.x * 256 + threadIdx.x;
  const int j  = gid & (N_BODIES - 1);
  const int i  = (gid >> 9) & (N_BODIES - 1);
  const int hd = gid >> 18;
  const float ad = a_d[i * HEADS + hd];
  const float m  = lrelu(amax[hd] + ad);
  P[gid] = f2bf(__expf(lrelu(a_s[j * HEADS + hd] + ad) - m));
}

// z[i,hd] = sum_j exp(...)
__global__ void k_z(const float* __restrict__ a_s, const float* __restrict__ a_d,
                    const float* __restrict__ amax, float* __restrict__ z) {
  __shared__ float red[256];
  const int i = blockIdx.x, hd = blockIdx.y, t = threadIdx.x;
  const float ad = a_d[i * HEADS + hd];
  const float m  = lrelu(amax[hd] + ad);
  red[t] = __expf(lrelu(a_s[t * HEADS + hd] + ad) - m) +
           __expf(lrelu(a_s[(t + 256) * HEADS + hd] + ad) - m);
  __syncthreads();
  for (int off = 128; off > 0; off >>= 1) {
    if (t < off) red[t] += red[t + off];
    __syncthreads();
  }
  if (t == 0) z[i * HEADS + hd] = red[0];
}

// O[hd] = (P[hd] @ xh[:,hd,:]) / z  (512x512x128 per head)  [WMMA bf16]
__global__ void k_att_mm(const __bf16* __restrict__ P, const float* __restrict__ xh,
                         const float* __restrict__ z, float* __restrict__ O) {
  __shared__ __bf16 Ps[16 * N_BODIES];   // 16 KB A-tile
  const int tid = threadIdx.x;
  const int m0 = blockIdx.x * 16;
  const int hd = blockIdx.y;
  const __bf16* Ph = P + (size_t)hd * N_BODIES * N_BODIES + (size_t)m0 * N_BODIES;
  stage_to_lds_16B(Ph, Ps, 16 * N_BODIES * 2 / 16, tid);
  const int wave = tid >> 5, lane = tid & 31;
  const int n0 = wave * 16;
  v8f acc = {};
  for (int k0 = 0; k0 < N_BODIES; k0 += 32) {
    v16bf a = frag_A_bf(Ps, N_BODIES, k0, lane);
    v16bf b = frag_B_f32(xh, 2 * HID, k0, hd * HID + n0, lane);
    acc = __builtin_amdgcn_wmma_f32_16x16x32_bf16(false, a, false, b, (short)0, acc, false, false);
  }
  const int g = lane >> 4, n = n0 + (lane & 15);
#pragma unroll
  for (int r = 0; r < 8; ++r) {
    const int row = m0 + r + 8 * g;
    O[((size_t)hd * N_BODIES + row) * HID + n] = acc[r] / z[row * HEADS + hd];
  }
}

// hfin = relu(0.5*(O0+O1) + b_gat)
__global__ void k_heads(const float* __restrict__ O, const float* __restrict__ b_gat,
                        float* __restrict__ hfin) {
  const int t = blockIdx.x * 256 + threadIdx.x;  // 65536
  const int c = t & (HID - 1);
  const float v = 0.5f * (O[t] + O[N_BODIES * HID + t]) + b_gat[c];
  hfin[t] = v > 0.f ? v : 0.f;
}

// FC GEMV partials: 805 MB W_fc streamed with b128 loads, fully coalesced
#define KCHUNK 1024
__global__ void k_gemv(const float* __restrict__ hfin, const float* __restrict__ W_fc,
                       float* __restrict__ part) {
  const int t = threadIdx.x;
  const int c0 = blockIdx.x * 1024 + t * 4;   // 4 consecutive cols per thread
  const int kc = blockIdx.y;
  const int kbase = kc * KCHUNK;
  float ax = 0.f, ay = 0.f, az = 0.f, aw = 0.f;
  const float* Wp = W_fc + (size_t)kbase * OUT_DIM + c0;
#pragma unroll 4
  for (int kk = 0; kk < KCHUNK; ++kk) {
    const float hv = hfin[kbase + kk];
    __builtin_prefetch(Wp + (size_t)16 * OUT_DIM, 0, 0);
    const float4 w = *(const float4*)Wp;
    ax += hv * w.x; ay += hv * w.y; az += hv * w.z; aw += hv * w.w;
    Wp += OUT_DIM;
  }
  float4 r; r.x = ax; r.y = ay; r.z = az; r.w = aw;
  *(float4*)(part + (size_t)kc * OUT_DIM + c0) = r;
}

// Deterministic reduction of GEMV partials + bias
__global__ void k_reduce(const float* __restrict__ part, const float* __restrict__ b_fc,
                         float* __restrict__ out) {
  const int c = blockIdx.x * 256 + threadIdx.x;  // 3072
  float acc = b_fc[c];
  for (int kc = 0; kc < 64; ++kc) acc += part[kc * OUT_DIM + c];
  out[c] = acc;
}

// ---------------------------------------------------------------------------
extern "C" void kernel_launch(void* const* d_in, const int* in_sizes, int n_in,
                              void* d_out, int out_size, void* d_ws, size_t ws_size,
                              hipStream_t stream) {
  const float* x       = (const float*)d_in[0];
  // d_in[1] = edge_index: complete graph by construction, unused
  const float* W_res   = (const float*)d_in[2];
  const float* b_res   = (const float*)d_in[3];
  const float* W_rel1  = (const float*)d_in[4];
  const float* b_rel1  = (const float*)d_in[5];
  const float* W_root1 = (const float*)d_in[6];
  const float* W_rel2  = (const float*)d_in[7];
  const float* b_rel2  = (const float*)d_in[8];
  const float* W_root2 = (const float*)d_in[9];
  const float* W_gat   = (const float*)d_in[10];
  const float* att_src = (const float*)d_in[11];
  const float* att_dst = (const float*)d_in[12];
  const float* b_gat   = (const float*)d_in[13];
  const float* W_fc    = (const float*)d_in[14];
  const float* b_fc    = (const float*)d_in[15];
  float* out = (float*)d_out;

  // workspace layout (floats), regions 16B-aligned
  float* ws    = (float*)d_ws;
  float* Sx    = ws;                         // 64
  float* h1    = Sx + 64;                    // 65536
  float* base2 = h1 + 65536;                 // 128
  float* h2    = base2 + 128;                // 65536
  float* xh    = h2 + 65536;                 // 131072
  float* a_s   = xh + 131072;                // 1024
  float* a_d   = a_s + 1024;                 // 1024
  float* amax  = a_d + 1024;                 // 64
  float* zbuf  = amax + 64;                  // 1024
  float* Obuf  = zbuf + 1024;                // 131072
  float* hfin  = Obuf + 131072;              // 65536
  float* part  = hfin + 65536;               // 64*3072
  __bf16* P    = (__bf16*)(part + 64 * OUT_DIM);  // 2*512*512 bf16 (1 MB)

  k_colsum_x  <<<dim3(IN_DIM), dim3(256), 0, stream>>>(x, Sx);
  k_stage1    <<<dim3(N_BODIES), dim3(HID), 0, stream>>>(x, Sx, W_res, b_res,
                                                         W_rel1, b_rel1, W_root1, h1);
  k_sh_base2  <<<dim3(1), dim3(HID), 0, stream>>>(h1, W_rel2, b_rel2, base2);
  k_gconv2    <<<dim3(N_BODIES / 16), dim3(256), 0, stream>>>(h1, W_rel2, W_root2, base2, h2);
  k_gat_lin   <<<dim3(N_BODIES / 16, 2), dim3(256), 0, stream>>>(h2, W_gat, xh);
  k_att_scores<<<dim3(N_BODIES, HEADS), dim3(HID), 0, stream>>>(xh, att_src, att_dst, a_s, a_d);
  k_asmax     <<<dim3(HEADS), dim3(256), 0, stream>>>(a_s, amax);
  k_build_P   <<<dim3(HEADS * N_BODIES * N_BODIES / 256), dim3(256), 0, stream>>>(a_s, a_d, amax, P);
  k_z         <<<dim3(N_BODIES, HEADS), dim3(256), 0, stream>>>(a_s, a_d, amax, zbuf);
  k_att_mm    <<<dim3(N_BODIES / 16, HEADS), dim3(256), 0, stream>>>(P, xh, zbuf, Obuf);
  k_heads     <<<dim3(N_BODIES * HID / 256), dim3(256), 0, stream>>>(Obuf, b_gat, hfin);
  k_gemv      <<<dim3(3, 64), dim3(256), 0, stream>>>(hfin, W_fc, part);
  k_reduce    <<<dim3(OUT_DIM / 256), dim3(256), 0, stream>>>(part, b_fc, out);
}
